// HierVAE_42099269435463
// MI455X (gfx1250) — matrix operand order repaired and checked by reading, not compiled, not run on hardware
//
#include <hip/hip_runtime.h>

// ---------------- problem constants ----------------
#define B_   512
#define S_   120
#define V_   512
#define E_   128
#define H_   256
#define G_   1024   // 4*H (i,f,g,o)
#define BI_  512    // 2*H
#define LAT_ 224    // 128+64+32

typedef __bf16 bf16_t;
typedef __attribute__((ext_vector_type(16))) __bf16 v16bf;
typedef __attribute__((ext_vector_type(8)))  __bf16 v8bf;
typedef __attribute__((ext_vector_type(8)))  float  v8f;

// ---------------- scalar helpers ----------------
__device__ __forceinline__ bf16_t f2bf(float f) {
    union { float f; unsigned u; } a; a.f = f;
    unsigned r = a.u + 0x7FFFu + ((a.u >> 16) & 1u);   // RNE
    union { unsigned short s; bf16_t b; } o; o.s = (unsigned short)(r >> 16);
    return o.b;
}
__device__ __forceinline__ float bf2f(bf16_t h) {
    union { unsigned short s; bf16_t b; } i; i.b = h;
    union { unsigned u; float f; } o; o.u = ((unsigned)i.s) << 16;
    return o.f;
}
__device__ __forceinline__ float sigm_f(float x) { return 1.f / (1.f + __expf(-x)); }
__device__ __forceinline__ float tanh_f(float x) {
    x = fminf(15.f, fmaxf(-15.f, x));
    float e = __expf(2.f * x);
    return (e - 1.f) / (e + 1.f);
}

// ---------------- WMMA fragment loaders (CDNA5 layouts) ----------------
// A (16x32 bf16): row = lane&15; lane<16 holds K {k0..k0+7, k0+16..k0+23},
// lane>=16 holds K {k0+8..k0+15, k0+24..k0+31}  (ISA 7.12.2 table).
__device__ __forceinline__ v16bf load_a_frag(const bf16_t* __restrict__ base,
                                             long long ld, int k0) {
    const int lane = threadIdx.x & 31;
    const bf16_t* p = base + (long long)(lane & 15) * ld + k0 + ((lane >> 4) << 3);
    v8bf c0 = *(const v8bf*)(p);
    v8bf c1 = *(const v8bf*)(p + 16);
    v16bf r;
#pragma unroll
    for (int i = 0; i < 8; ++i) { r[i] = c0[i]; r[i + 8] = c1[i]; }
    return r;
}
// B (32x16 bf16, row-major KxN source): col = lane&15; lane<16 holds K {0..7, 16..23},
// lane>=16 holds K {8..15, 24..31} (same K order as A).
__device__ __forceinline__ v16bf load_b_frag(const bf16_t* __restrict__ Brow,
                                             int ldb, int n0) {
    const int lane = threadIdx.x & 31;
    const bf16_t* p = Brow + n0 + (lane & 15) + (long long)((lane >> 4) << 3) * ldb;
    v16bf r;
#pragma unroll
    for (int i = 0; i < 8; ++i) {
        r[i] = p[(long long)i * ldb];
        r[i + 8] = p[(long long)(16 + i) * ldb];
    }
    return r;
}
// C/D (16x16 f32): col = lane&15, row = vgpr + 8*(lane>=16).

// ---------------- fused LSTM timestep kernel ----------------
// gates = [x_t, h_{t-1}] @ Wcat + bias; elementwise LSTM; writes h (bf16) and
// optionally y (layer output, bf16).
// Each wave owns a 32x16 h-slab (2 M-tiles x 1 N-tile): it loads the 4 gate
// B-fragments once per k-iter and reuses them for both A-fragments (8 WMMAs
// per 4 b128 A loads + 64 2-byte B loads).  Waves of a block share n0 (B-frag addresses identical
// across the block -> cache hits); M-groups are distinct (no redundant A).
// Grid must be exactly (B/32)*(H/16)/8 = 32 blocks of 256 threads.
__global__ __launch_bounds__(256) void lstm_step_kernel(
    const bf16_t* __restrict__ xbase, long long x_ld, int in_w,
    const bf16_t* __restrict__ Wcat,        // (in_w+H) x G, bf16, row-major
    const float*  __restrict__ bias,        // G
    const bf16_t* __restrict__ h_in,        // B x H
    float*        __restrict__ c,           // B x H (fp32, read/write)
    bf16_t*       __restrict__ h_out,       // B x H
    bf16_t*       __restrict__ y,           // nullable; y + row*y_ld + col
    long long y_ld)
{
    const int lane = threadIdx.x & 31;
    const int wave = threadIdx.x >> 5;
    const int tile = blockIdx.x * 8 + wave;     // 256 tiles, n-major
    const int m0 = (tile & 15) << 5;            // 16 M-groups of 32 rows
    const int n0 = (tile >> 4) << 4;            // 16 N-tiles across H

    v8f acc[4][2];
#pragma unroll
    for (int g = 0; g < 4; ++g)
#pragma unroll
        for (int mt = 0; mt < 2; ++mt)
#pragma unroll
            for (int i = 0; i < 8; ++i) acc[g][mt][i] = 0.f;

    // phase 1: input contribution (K = in_w)
    const bf16_t* aX = xbase + (long long)m0 * x_ld;
    for (int k0 = 0; k0 < in_w; k0 += 32) {
        v16bf a0 = load_a_frag(aX, x_ld, k0);
        v16bf a1 = load_a_frag(aX + 16 * x_ld, x_ld, k0);
        const bf16_t* Brow = Wcat + (long long)k0 * G_;
        __builtin_prefetch(Brow + (long long)32 * G_, 0, 1);
#pragma unroll
        for (int g = 0; g < 4; ++g) {
            v16bf b = load_b_frag(Brow, G_, g * H_ + n0);
            acc[g][0] = __builtin_amdgcn_wmma_f32_16x16x32_bf16(
                false, a0, false, b, (short)0, acc[g][0], false, false);
            acc[g][1] = __builtin_amdgcn_wmma_f32_16x16x32_bf16(
                false, a1, false, b, (short)0, acc[g][1], false, false);
        }
    }
    // phase 2: recurrent contribution (K = H)
    const bf16_t* aH = h_in + (long long)m0 * H_;
    const bf16_t* Wh = Wcat + (long long)in_w * G_;
    for (int k0 = 0; k0 < H_; k0 += 32) {
        v16bf a0 = load_a_frag(aH, H_, k0);
        v16bf a1 = load_a_frag(aH + 16 * H_, H_, k0);
        const bf16_t* Brow = Wh + (long long)k0 * G_;
#pragma unroll
        for (int g = 0; g < 4; ++g) {
            v16bf b = load_b_frag(Brow, G_, g * H_ + n0);
            acc[g][0] = __builtin_amdgcn_wmma_f32_16x16x32_bf16(
                false, a0, false, b, (short)0, acc[g][0], false, false);
            acc[g][1] = __builtin_amdgcn_wmma_f32_16x16x32_bf16(
                false, a1, false, b, (short)0, acc[g][1], false, false);
        }
    }

    // epilogue: bias + LSTM pointwise
    const int col = lane & 15;
    const int rb  = (lane >> 4) << 3;
    const float bi = bias[0 * H_ + n0 + col];
    const float bf = bias[1 * H_ + n0 + col];
    const float bg = bias[2 * H_ + n0 + col];
    const float bo = bias[3 * H_ + n0 + col];
#pragma unroll
    for (int mt = 0; mt < 2; ++mt) {
#pragma unroll
        for (int v = 0; v < 8; ++v) {
            const int row = m0 + mt * 16 + rb + v;
            const long long cidx = (long long)row * H_ + n0 + col;
            float iv = acc[0][mt][v] + bi;
            float fv = acc[1][mt][v] + bf;
            float gv = acc[2][mt][v] + bg;
            float ov = acc[3][mt][v] + bo;
            float cn = sigm_f(fv) * c[cidx] + sigm_f(iv) * tanh_f(gv);
            c[cidx] = cn;
            float hv = sigm_f(ov) * tanh_f(cn);
            bf16_t hb = f2bf(hv);
            h_out[cidx] = hb;
            if (y) y[(long long)row * y_ld + n0 + col] = hb;
        }
    }
}

// ---------------- generic WMMA GEMM: C = A(bf16 512xK) @ Bt(bf16 KxN) + bias
// Each wave: 32x16 slab (2 M-tiles).  Tiles n-major so block waves share B.
// Grid must be exactly (512/32)*ntiles_n/8 = 2*ntiles_n blocks of 256.
__global__ __launch_bounds__(256) void wmma_gemm_kernel(
    const bf16_t* __restrict__ A, long long lda,
    const bf16_t* __restrict__ Bt, int ldb,
    const float*  __restrict__ bias,
    float*        __restrict__ C, long long ldc,
    int K)
{
    const int lane = threadIdx.x & 31;
    const int tile = blockIdx.x * 8 + (threadIdx.x >> 5);
    const int m0 = (tile & 15) << 5;            // 16 M-groups of 32 rows
    const int n0 = (tile >> 4) << 4;

    v8f acc[2];
#pragma unroll
    for (int mt = 0; mt < 2; ++mt)
#pragma unroll
        for (int i = 0; i < 8; ++i) acc[mt][i] = 0.f;

    const bf16_t* aB = A + (long long)m0 * lda;
    for (int k0 = 0; k0 < K; k0 += 32) {
        v16bf a0 = load_a_frag(aB, lda, k0);
        v16bf a1 = load_a_frag(aB + 16 * lda, lda, k0);
        v16bf b = load_b_frag(Bt + (long long)k0 * ldb, ldb, n0);
        acc[0] = __builtin_amdgcn_wmma_f32_16x16x32_bf16(
            false, a0, false, b, (short)0, acc[0], false, false);
        acc[1] = __builtin_amdgcn_wmma_f32_16x16x32_bf16(
            false, a1, false, b, (short)0, acc[1], false, false);
    }
    const int col = lane & 15;
    const int rb  = (lane >> 4) << 3;
    const float bb = bias ? bias[n0 + col] : 0.f;
#pragma unroll
    for (int mt = 0; mt < 2; ++mt)
#pragma unroll
        for (int v = 0; v < 8; ++v)
            C[(long long)(m0 + mt * 16 + rb + v) * ldc + n0 + col] = acc[mt][v] + bb;
}

// ---------------- weight prep: Wcat[k,n] = k<in ? Wih[n,k] : Whh[n,k-in] (bf16)
__global__ void prep_wcat_kernel(const float* __restrict__ Wih,
                                 const float* __restrict__ Whh,
                                 bf16_t* __restrict__ out,
                                 int in_w, int hdim, int ncols, int total)
{
    int idx = blockIdx.x * blockDim.x + threadIdx.x;
    if (idx >= total) return;
    int k = idx / ncols, n = idx % ncols;
    float v = (k < in_w) ? Wih[(long long)n * in_w + k]
                         : Whh[(long long)n * hdim + (k - in_w)];
    out[idx] = f2bf(v);
}

// ---------------- embedding gathers ----------------
__global__ void embed_seq_kernel(const int* __restrict__ x,
                                 const float* __restrict__ emb,
                                 bf16_t* __restrict__ out, int total)
{
    int idx = blockIdx.x * blockDim.x + threadIdx.x;
    if (idx >= total) return;
    int e = idx & (E_ - 1);
    int bs = idx >> 7;                 // E_ == 128
    out[idx] = f2bf(emb[(long long)x[bs] * E_ + e]);
}
__global__ void embed_tok_kernel(const int* __restrict__ tok,
                                 const float* __restrict__ emb,
                                 bf16_t* __restrict__ out, int total)
{
    int idx = blockIdx.x * blockDim.x + threadIdx.x;
    if (idx >= total) return;
    int e = idx & (E_ - 1);
    int b = idx >> 7;
    out[idx] = f2bf(emb[(long long)tok[b] * E_ + e]);
}

// ---------------- final-h extraction: hcat = [y[:,S-1,:H] ; y[:,0,H:2H]] fp32
__global__ void hcat_extract_kernel(const bf16_t* __restrict__ y,
                                    float* __restrict__ hcat, int total)
{
    int idx = blockIdx.x * blockDim.x + threadIdx.x;
    if (idx >= total) return;
    int b = idx / BI_, j = idx % BI_;
    bf16_t v = (j < H_) ? y[(long long)b * S_ * BI_ + (long long)(S_ - 1) * BI_ + j]
                        : y[(long long)b * S_ * BI_ + j];
    hcat[idx] = bf2f(v);
}

// ---------------- small fp32 linear: Y[m,n] = X[m,:]·W[n,:] + b[n] ----------
__global__ void linear_f32_kernel(const float* __restrict__ X, int K,
                                  const float* __restrict__ W,
                                  const float* __restrict__ b,
                                  float* __restrict__ Y, int N, int total)
{
    int idx = blockIdx.x * blockDim.x + threadIdx.x;
    if (idx >= total) return;
    int m = idx / N, n = idx % N;
    const float* xr = X + (long long)m * K;
    const float* wr = W + (long long)n * K;
    float s = b[n];
    for (int k = 0; k < K; ++k) s += xr[k] * wr[k];
    Y[idx] = s;
}

// ---------------- reparameterize: z = mu + eps * exp(0.5*lv) ----------------
__global__ void z_kernel(const float* mu0, const float* lv0,
                         const float* mu1, const float* lv1,
                         const float* mu2, const float* lv2,
                         const float* __restrict__ eps, float* __restrict__ z,
                         int total)
{
    int idx = blockIdx.x * blockDim.x + threadIdx.x;
    if (idx >= total) return;
    int b = idx / LAT_, d = idx % LAT_;
    float mu, lv;
    if (d < 128)      { mu = mu0[b * 128 + d];        lv = lv0[b * 128 + d]; }
    else if (d < 192) { mu = mu1[b * 64 + d - 128];   lv = lv1[b * 64 + d - 128]; }
    else              { mu = mu2[b * 32 + d - 192];   lv = lv2[b * 32 + d - 192]; }
    z[idx] = mu + eps[idx] * __expf(0.5f * lv);
}

// ---------------- decoder init: replicate h_init (bf16) into 3 layer states --
__global__ void dec_init_kernel(const float* __restrict__ h_init,
                                bf16_t* h0, bf16_t* h1, bf16_t* h2, int total)
{
    int idx = blockIdx.x * blockDim.x + threadIdx.x;
    if (idx >= total) return;
    bf16_t v = f2bf(h_init[idx]);
    h0[idx] = v; h1[idx] = v; h2[idx] = v;
}

// ---------------- argmax over V=512 logits, one wave per row ----------------
__global__ __launch_bounds__(256) void argmax_kernel(
    const float* __restrict__ logits, long long ld, int* __restrict__ tok)
{
    int b = blockIdx.x * 8 + (threadIdx.x >> 5);
    int lane = threadIdx.x & 31;
    const float* row = logits + (long long)b * ld;
    float best = -3.4e38f; int bi = 0;
    for (int j = lane; j < V_; j += 32) {
        float v = row[j];
        if (v > best) { best = v; bi = j; }
    }
#pragma unroll
    for (int off = 16; off; off >>= 1) {
        float ov = __shfl_down(best, off, 32);
        int   oi = __shfl_down(bi,   off, 32);
        if (ov > best || (ov == best && oi < bi)) { best = ov; bi = oi; }
    }
    if (lane == 0) tok[b] = bi;
}

// =======================================================================
extern "C" void kernel_launch(void* const* d_in, const int* in_sizes, int n_in,
                              void* d_out, int out_size, void* d_ws, size_t ws_size,
                              hipStream_t stream)
{
    (void)in_sizes; (void)n_in; (void)out_size; (void)ws_size;

    const int*   x       = (const int*)  d_in[0];
    const float* eps     = (const float*)d_in[1];
    const float* emb     = (const float*)d_in[2];
    const float* e0_Wih  = (const float*)d_in[3];
    const float* e0_Whh  = (const float*)d_in[4];
    const float* e0_b    = (const float*)d_in[5];
    const float* enc_Wih = (const float*)d_in[6];
    const float* enc_Whh = (const float*)d_in[7];
    const float* enc_b   = (const float*)d_in[8];
    const float* mu_W[3] = {(const float*)d_in[9],  (const float*)d_in[13], (const float*)d_in[17]};
    const float* mu_b[3] = {(const float*)d_in[10], (const float*)d_in[14], (const float*)d_in[18]};
    const float* lv_W[3] = {(const float*)d_in[11], (const float*)d_in[15], (const float*)d_in[19]};
    const float* lv_b[3] = {(const float*)d_in[12], (const float*)d_in[16], (const float*)d_in[20]};
    const float* di_W    = (const float*)d_in[21];
    const float* di_b    = (const float*)d_in[22];
    const float* d0_Wih  = (const float*)d_in[23];
    const float* d0_Whh  = (const float*)d_in[24];
    const float* d0_b    = (const float*)d_in[25];
    const float* d_Wih   = (const float*)d_in[26];
    const float* d_Whh   = (const float*)d_in[27];
    const float* d_b     = (const float*)d_in[28];
    const float* fc_W    = (const float*)d_in[29];
    const float* fc_b    = (const float*)d_in[30];
    float* out = (float*)d_out;

    // ---- output layout: recon (B,S,V) then mu0,lv0,mu1,lv1,mu2,lv2 ----
    float* recon = out;
    long long o = (long long)B_ * S_ * V_;
    float* mu0 = out + o; o += B_ * 128;
    float* lv0 = out + o; o += B_ * 128;
    float* mu1 = out + o; o += B_ * 64;
    float* lv1 = out + o; o += B_ * 64;
    float* mu2 = out + o; o += B_ * 32;
    float* lv2 = out + o;

    // ---- workspace carve (~158 MB) ----
    char* wsp = (char*)d_ws; size_t off = 0;
    auto take = [&](size_t bytes) -> void* {
        void* r = wsp + off; off += (bytes + 255) & ~(size_t)255; return r;
    };
    bf16_t* xA      = (bf16_t*)take((size_t)B_ * S_ * BI_ * 2);
    bf16_t* xB      = (bf16_t*)take((size_t)B_ * S_ * BI_ * 2);
    bf16_t* Wc_e0   = (bf16_t*)take((size_t)2 * (E_ + H_) * G_ * 2);
    bf16_t* Wc_enc  = (bf16_t*)take((size_t)16 * (BI_ + H_) * G_ * 2);
    bf16_t* Wc_d0   = (bf16_t*)take((size_t)(E_ + H_) * G_ * 2);
    bf16_t* Wc_d12  = (bf16_t*)take((size_t)2 * (2 * H_) * G_ * 2);
    bf16_t* fcWT    = (bf16_t*)take((size_t)H_ * V_ * 2);
    bf16_t* hEa     = (bf16_t*)take((size_t)B_ * H_ * 2);
    bf16_t* hEb     = (bf16_t*)take((size_t)B_ * H_ * 2);
    float*  c_enc   = (float*) take((size_t)B_ * H_ * 4);
    bf16_t* hdP[3], *hdN[3]; float* c_dec[3];
    for (int l = 0; l < 3; ++l) {
        hdP[l]   = (bf16_t*)take((size_t)B_ * H_ * 2);
        hdN[l]   = (bf16_t*)take((size_t)B_ * H_ * 2);
        c_dec[l] = (float*) take((size_t)B_ * H_ * 4);
    }
    float* hcat[3];
    for (int s = 0; s < 3; ++s) hcat[s] = (float*)take((size_t)B_ * BI_ * 4);
    float*  zbuf   = (float*) take((size_t)B_ * LAT_ * 4);
    float*  h_init = (float*) take((size_t)B_ * H_ * 4);
    bf16_t* x_dec  = (bf16_t*)take((size_t)B_ * E_ * 2);
    int*    tok    = (int*)   take((size_t)B_ * 4);

    auto blocks = [](long long n) { return (unsigned)((n + 255) / 256); };

    // ---- one-time (per launch) weight packing to bf16 WMMA layout ----
    for (int d = 0; d < 2; ++d)
        prep_wcat_kernel<<<blocks((E_ + H_) * G_), 256, 0, stream>>>(
            e0_Wih + (size_t)d * G_ * E_, e0_Whh + (size_t)d * G_ * H_,
            Wc_e0 + (size_t)d * (E_ + H_) * G_, E_, H_, G_, (E_ + H_) * G_);
    for (int ld8 = 0; ld8 < 16; ++ld8)
        prep_wcat_kernel<<<blocks((BI_ + H_) * G_), 256, 0, stream>>>(
            enc_Wih + (size_t)ld8 * G_ * BI_, enc_Whh + (size_t)ld8 * G_ * H_,
            Wc_enc + (size_t)ld8 * (BI_ + H_) * G_, BI_, H_, G_, (BI_ + H_) * G_);
    prep_wcat_kernel<<<blocks((E_ + H_) * G_), 256, 0, stream>>>(
        d0_Wih, d0_Whh, Wc_d0, E_, H_, G_, (E_ + H_) * G_);
    for (int l = 0; l < 2; ++l)
        prep_wcat_kernel<<<blocks(2 * H_ * G_), 256, 0, stream>>>(
            d_Wih + (size_t)l * G_ * H_, d_Whh + (size_t)l * G_ * H_,
            Wc_d12 + (size_t)l * (2 * H_) * G_, H_, H_, G_, 2 * H_ * G_);
    prep_wcat_kernel<<<blocks(H_ * V_), 256, 0, stream>>>(
        fc_W, fc_W, fcWT, 0, H_, V_, H_ * V_);   // pure transpose of fc_W

    // ---- embedding of input sequence ----
    embed_seq_kernel<<<blocks((long long)B_ * S_ * E_), 256, 0, stream>>>(
        x, emb, xA, B_ * S_ * E_);

    // ---- 9 bidirectional LSTM layers (3 stacks of 3) ----
    bf16_t* bufs[2] = {xA, xB};
    for (int l = 0; l < 9; ++l) {
        const int in_w = (l == 0) ? E_ : BI_;
        bf16_t* xin  = bufs[l & 1];
        bf16_t* xout = bufs[(l + 1) & 1];
        for (int dir = 0; dir < 2; ++dir) {
            const bf16_t* Wc = (l == 0)
                ? Wc_e0 + (size_t)dir * (E_ + H_) * G_
                : Wc_enc + ((size_t)(l - 1) * 2 + dir) * (BI_ + H_) * G_;
            const float* bias = (l == 0) ? e0_b + (size_t)dir * G_
                                         : enc_b + ((size_t)(l - 1) * 2 + dir) * G_;
            hipMemsetAsync(hEa,  0, (size_t)B_ * H_ * 2, stream);
            hipMemsetAsync(c_enc, 0, (size_t)B_ * H_ * 4, stream);
            bf16_t *hin = hEa, *hout = hEb;
            for (int s = 0; s < S_; ++s) {
                const int t = dir ? (S_ - 1 - s) : s;
                lstm_step_kernel<<<32, 256, 0, stream>>>(
                    xin + (long long)t * in_w, (long long)S_ * in_w, in_w,
                    Wc, bias, hin, c_enc, hout,
                    xout + (long long)t * BI_ + dir * H_, (long long)S_ * BI_);
                bf16_t* tmp = hin; hin = hout; hout = tmp;
            }
        }
        if (l == 2) hcat_extract_kernel<<<blocks(B_ * BI_), 256, 0, stream>>>(xout, hcat[0], B_ * BI_);
        if (l == 5) hcat_extract_kernel<<<blocks(B_ * BI_), 256, 0, stream>>>(xout, hcat[1], B_ * BI_);
        if (l == 8) hcat_extract_kernel<<<blocks(B_ * BI_), 256, 0, stream>>>(xout, hcat[2], B_ * BI_);
    }

    // ---- latent heads + reparameterize + decoder init ----
    const int lat[3] = {128, 64, 32};
    float* muP[3] = {mu0, mu1, mu2};
    float* lvP[3] = {lv0, lv1, lv2};
    for (int s = 0; s < 3; ++s) {
        linear_f32_kernel<<<blocks(B_ * lat[s]), 256, 0, stream>>>(
            hcat[s], BI_, mu_W[s], mu_b[s], muP[s], lat[s], B_ * lat[s]);
        linear_f32_kernel<<<blocks(B_ * lat[s]), 256, 0, stream>>>(
            hcat[s], BI_, lv_W[s], lv_b[s], lvP[s], lat[s], B_ * lat[s]);
    }
    z_kernel<<<blocks(B_ * LAT_), 256, 0, stream>>>(mu0, lv0, mu1, lv1, mu2, lv2,
                                                    eps, zbuf, B_ * LAT_);
    linear_f32_kernel<<<blocks(B_ * H_), 256, 0, stream>>>(
        zbuf, LAT_, di_W, di_b, h_init, H_, B_ * H_);
    dec_init_kernel<<<blocks(B_ * H_), 256, 0, stream>>>(
        h_init, hdP[0], hdP[1], hdP[2], B_ * H_);
    for (int l = 0; l < 3; ++l)
        hipMemsetAsync(c_dec[l], 0, (size_t)B_ * H_ * 4, stream);
    hipMemsetAsync(tok, 0, (size_t)B_ * 4, stream);

    // ---- autoregressive greedy decode, 120 steps ----
    for (int t = 0; t < S_; ++t) {
        embed_tok_kernel<<<blocks(B_ * E_), 256, 0, stream>>>(tok, emb, x_dec, B_ * E_);
        lstm_step_kernel<<<32, 256, 0, stream>>>(
            x_dec, (long long)E_, E_, Wc_d0, d0_b,
            hdP[0], c_dec[0], hdN[0], (bf16_t*)nullptr, 0);
        lstm_step_kernel<<<32, 256, 0, stream>>>(
            hdN[0], (long long)H_, H_, Wc_d12, d_b,
            hdP[1], c_dec[1], hdN[1], (bf16_t*)nullptr, 0);
        lstm_step_kernel<<<32, 256, 0, stream>>>(
            hdN[1], (long long)H_, H_, Wc_d12 + (size_t)(2 * H_) * G_, d_b + G_,
            hdP[2], c_dec[2], hdN[2], (bf16_t*)nullptr, 0);
        // logits -> d_out[:, t, :]   (row stride S*V); N tiles = 32 -> 64 blocks
        wmma_gemm_kernel<<<64, 256, 0, stream>>>(
            hdN[2], (long long)H_, fcWT, V_, fc_b,
            recon + (long long)t * V_, (long long)S_ * V_, H_);
        argmax_kernel<<<64, 256, 0, stream>>>(recon + (long long)t * V_,
                                              (long long)S_ * V_, tok);
        for (int l = 0; l < 3; ++l) { bf16_t* tmp = hdP[l]; hdP[l] = hdN[l]; hdN[l] = tmp; }
    }
}
